// LSTM_DOUBLE_ATT_STACK_PARALLEL_MUL_OUT_57990648430774
// MI455X (gfx1250) — matrix operand order repaired and checked by reading, not compiled, not run on hardware
//
#include <hip/hip_runtime.h>
#include <stdint.h>

// ---------------------------------------------------------------------------
// LSTM_DOUBLE_ATT_STACK_PARALLEL for MI455X (gfx1250, wave32, WMMA).
// All large GEMMs run on v_wmma_f32_16x16x32_f16 (f16 A/B, f32 accum).
// The [B,A,A] pv tensor (157MB) is never materialized: soft-attention is
// fully fused (WMMA -> tanh -> dot(d2d) -> softmax in LDS -> re-weight att).
// a2a_w staged in 204KB of LDS (CDNA5 320KB/WGP), pitch 520 f16 so B-frag
// lanes hit banks 0,4,...,60 conflict-free.
// The attention column loop is a RUNTIME loop over pairs of 16-col tiles:
// runtime addressing stops LICM from hoisting the (mt-invariant) LDS loads
// out of the row loop (the cause of the scratch spills in earlier builds),
// while the two accumulator chains still overlap ds_load latency with WMMA.
// Workspace use: ~294 MB (att f16 copy dominates at 206 MB).
// ---------------------------------------------------------------------------

typedef __attribute__((ext_vector_type(16))) _Float16 v16h;
typedef __attribute__((ext_vector_type(8)))  _Float16 v8h;
typedef __attribute__((ext_vector_type(4)))  _Float16 v4h;
typedef __attribute__((ext_vector_type(8)))  float    v8f;

union V16U { v16h v; v8h h[2]; };

__device__ __forceinline__ v8f wmma16(v16h a, v16h b, v8f c) {
  return __builtin_amdgcn_wmma_f32_16x16x32_f16(false, a, false, b, (short)0, c,
                                                false, false);
}

// A-fragment (16x32 f16, row-major src, ld=512). base = row*512 + half*16.
// lanes 0-15: elems 0..15 = K+0..15
// lanes16-31: elems 0..15 = K+16..31
__device__ __forceinline__ v16h load_a16(const _Float16* base, int k32) {
  V16U u;
  u.h[0] = *(const v8h*)(base + k32);
  u.h[1] = *(const v8h*)(base + k32 + 8);
  return u.v;
}
// B-fragment (32x16 f16). W stored [N][K] row-major (K contiguous).
// base = col*pitch + half*16. lanes 0-15: K k0..k0+15; lanes16-31: k0+16..31.
__device__ __forceinline__ v16h load_b16(const _Float16* base, int k32) {
  V16U u;
  u.h[0] = *(const v8h*)(base + k32);
  u.h[1] = *(const v8h*)(base + k32 + 8);
  return u.v;
}

// ---------------------------------------------------------------------------
// f32 -> f16 conversion (vectorized x4, grid-stride)
// ---------------------------------------------------------------------------
__global__ __launch_bounds__(256) void cvt_f16_kernel(
    const float* __restrict__ s, _Float16* __restrict__ d, long long n4) {
  long long i = (long long)blockIdx.x * blockDim.x + threadIdx.x;
  long long stride = (long long)gridDim.x * blockDim.x;
  for (; i < n4; i += stride) {
    float4 f = ((const float4*)s)[i];
    v4h h = {(_Float16)f.x, (_Float16)f.y, (_Float16)f.z, (_Float16)f.w};
    ((v4h*)d)[i] = h;
  }
}

// ---------------------------------------------------------------------------
// Fused soft-attention. One block per batch element b.
//   ph[a]    = h[b] . h2a_w[a] + h2a_b[a]                       (VALU, LDS)
//   pv[a,k]  = att[b,a,:] . a2a_w[k,:]                          (WMMA f16)
//   score[a] = sum_k tanh(pv+ph[a]+a2a_b[k]) * d2d_w[k]         (fused)
//   w = softmax(score); out[r] = sum_a w[a]*att32[b,a,r]
// A padded 196 -> 13 tiles of 16; invalid rows/cols masked at accumulation.
// ---------------------------------------------------------------------------
#define BPITCH 520
#define SATT_LDS_BYTES (196 * BPITCH * 2)

__global__ __launch_bounds__(256) void soft_att_kernel(
    const _Float16* __restrict__ att16, const float* __restrict__ att32,
    const float* __restrict__ h, const _Float16* __restrict__ a2aw,
    const float* __restrict__ a2ab, const float* __restrict__ h2aw,
    const float* __restrict__ h2ab, const float* __restrict__ d2dw,
    float* __restrict__ out32, _Float16* __restrict__ out16) {
  extern __shared__ _Float16 s_b[];  // [196][BPITCH] staged a2a_w
  __shared__ float h_s[512];
  __shared__ float ph_s[208];
  __shared__ float score_s[208];
  __shared__ float red_s[256];
  __shared__ float w_s[208];

  const int b = blockIdx.x, t = threadIdx.x;
  const _Float16* attb16 = att16 + (size_t)b * (196 * 512);
  const float*    attb32 = att32 + (size_t)b * (196 * 512);

  // stage a2a_w -> LDS (coalesced 16B chunks; 196 rows * 64 chunks)
  for (int idx = t; idx < 196 * 64; idx += 256) {
    const int row = idx >> 6, ch = idx & 63;
    const uint4 v = ((const uint4*)a2aw)[idx];
    *(uint4*)(s_b + row * BPITCH + ch * 8) = v;
  }

  h_s[t] = h[b * 512 + t];
  h_s[t + 256] = h[b * 512 + 256 + t];
  if (t < 208) score_s[t] = 0.f;
  __syncthreads();

  if (t < 196) {
    float acc = h2ab[t];
    const float4* wr4 = (const float4*)(h2aw + t * 512);
    for (int r4 = 0; r4 < 128; ++r4) {
      const float4 wv = wr4[r4];
      acc = fmaf(h_s[r4 * 4 + 0], wv.x, acc);
      acc = fmaf(h_s[r4 * 4 + 1], wv.y, acc);
      acc = fmaf(h_s[r4 * 4 + 2], wv.z, acc);
      acc = fmaf(h_s[r4 * 4 + 3], wv.w, acc);
    }
    ph_s[t] = acc;
  } else if (t < 208) {
    ph_s[t] = 0.f;
  }
  __syncthreads();

  const int wave = t >> 5, lane = t & 31;
  const int colT = lane & 15, half = lane >> 4;

  for (int mt = wave; mt < 13; mt += 8) {
    const int m0 = mt * 16;
    int arow = m0 + colT;
    if (arow > 195) arow = 195;  // clamp (garbage rows masked below)
    const _Float16* ab = attb16 + arow * 512 + half * 16;

    float rowacc[8] = {0.f, 0.f, 0.f, 0.f, 0.f, 0.f, 0.f, 0.f};

    // RUNTIME loop over pairs of column tiles: keeps LDS load addresses
    // non-hoistable and register pressure bounded (~130 VGPRs).
#pragma unroll 1
    for (int nt = 0; nt < 13; nt += 2) {
      const int colA = nt * 16 + colT;
      const int colB = colA + 16;  // tile 13 (last pair) is a masked dummy
      const int colAc = (colA > 195) ? 195 : colA;
      const int colBc = (colB > 195) ? 195 : colB;
      const _Float16* bbA = s_b + colAc * BPITCH + half * 16;
      const _Float16* bbB = s_b + colBc * BPITCH + half * 16;

      v8f cA = {}, cB = {};
#pragma unroll
      for (int kk = 0; kk < 2; ++kk) {
        v16h afr[8];
#pragma unroll
        for (int k = 0; k < 8; ++k) afr[k] = load_a16(ab, (kk * 8 + k) * 32);
#pragma unroll
        for (int k = 0; k < 8; ++k) {
          const int ko = (kk * 8 + k) * 32;
          cA = wmma16(afr[k], load_b16(bbA, ko), cA);
          cB = wmma16(afr[k], load_b16(bbB, ko), cB);
        }
      }

      const float dA = (colA <= 195) ? d2dw[colA] : 0.f;
      const float aA = (colA <= 195) ? a2ab[colA] : 0.f;
      const float dB = (colB <= 195) ? d2dw[colB] : 0.f;
      const float aB = (colB <= 195) ? a2ab[colB] : 0.f;
#pragma unroll
      for (int j = 0; j < 8; ++j) {
        const float ph = ph_s[m0 + j + half * 8];
        rowacc[j] += tanhf(cA[j] + ph + aA) * dA + tanhf(cB[j] + ph + aB) * dB;
      }
    }

#pragma unroll
    for (int j = 0; j < 8; ++j) {
      const int rr = m0 + j + half * 8;
      if (rr < 196) atomicAdd(&score_s[rr], rowacc[j]);
    }
  }
  __syncthreads();

  // softmax over score_s[0..195]
  const float sv = (t < 196) ? score_s[t] : -3.4e38f;
  red_s[t] = sv;
  __syncthreads();
  for (int s = 128; s > 0; s >>= 1) {
    if (t < s) red_s[t] = fmaxf(red_s[t], red_s[t + s]);
    __syncthreads();
  }
  const float mx = red_s[0];
  __syncthreads();
  const float ev = (t < 196) ? __expf(sv - mx) : 0.f;
  red_s[t] = ev;
  __syncthreads();
  for (int s = 128; s > 0; s >>= 1) {
    if (t < s) red_s[t] += red_s[t + s];
    __syncthreads();
  }
  if (t < 196) w_s[t] = ev / red_s[0];
  __syncthreads();

  // out[r] = sum_a w[a] * att32[b,a,r]  (fp32 for output accuracy, coalesced)
  float o0 = 0.f, o1 = 0.f;
  for (int a = 0; a < 196; ++a) {
    const float wa = w_s[a];
    o0 = fmaf(wa, attb32[a * 512 + t], o0);
    o1 = fmaf(wa, attb32[a * 512 + 256 + t], o1);
  }
  out32[(size_t)b * 512 + t] = o0;
  out32[(size_t)b * 512 + 256 + t] = o1;
  out16[(size_t)b * 512 + t] = (_Float16)o0;
  out16[(size_t)b * 512 + 256 + t] = (_Float16)o1;
}

// ---------------------------------------------------------------------------
// Cores pre-activation GEMM: sums[b, p*2048+k4] =
//   xt.i2h + prev_h.h2h + att_res.a2h   (biases added in epilogue)
// M=1024, N=4096, K=512x3 sources. Block = 16 rows x 512 cols, grid (64,8).
// A-chunk (8 frags) resident per source/K-half; 4 independent B/WMMA chains.
// ---------------------------------------------------------------------------
__global__ __launch_bounds__(256) void cores_gemm_kernel(
    const _Float16* __restrict__ Ax, const _Float16* __restrict__ Ah,
    const _Float16* __restrict__ Aa, const _Float16* __restrict__ Wx,
    const _Float16* __restrict__ Wh, const _Float16* __restrict__ Wa,
    float* __restrict__ sums) {
  const int wave = threadIdx.x >> 5, lane = threadIdx.x & 31;
  const int colT = lane & 15, half = lane >> 4;
  const int m0 = blockIdx.x * 16;
  const int n0 = blockIdx.y * 512 + wave * 64;
  const int aoff = (m0 + colT) * 512 + half * 16;

  const size_t b0 = (size_t)(n0 + 0 + colT) * 512 + half * 16;
  const size_t b1 = (size_t)(n0 + 16 + colT) * 512 + half * 16;
  const size_t b2 = (size_t)(n0 + 32 + colT) * 512 + half * 16;
  const size_t b3 = (size_t)(n0 + 48 + colT) * 512 + half * 16;

  const _Float16* As[3]  = {Ax + aoff, Ah + aoff, Aa + aoff};
  const _Float16* B0s[3] = {Wx + b0, Wh + b0, Wa + b0};
  const _Float16* B1s[3] = {Wx + b1, Wh + b1, Wa + b1};
  const _Float16* B2s[3] = {Wx + b2, Wh + b2, Wa + b2};
  const _Float16* B3s[3] = {Wx + b3, Wh + b3, Wa + b3};

  v8f c0 = {}, c1 = {}, c2 = {}, c3 = {};
#pragma unroll
  for (int s = 0; s < 3; ++s) {
#pragma unroll
    for (int kk = 0; kk < 2; ++kk) {
      v16h afr[8];
#pragma unroll
      for (int k = 0; k < 8; ++k) afr[k] = load_a16(As[s], (kk * 8 + k) * 32);
#pragma unroll
      for (int k = 0; k < 8; ++k) {
        const int ko = (kk * 8 + k) * 32;
        c0 = wmma16(afr[k], load_b16(B0s[s], ko), c0);
        c1 = wmma16(afr[k], load_b16(B1s[s], ko), c1);
        c2 = wmma16(afr[k], load_b16(B2s[s], ko), c2);
        c3 = wmma16(afr[k], load_b16(B3s[s], ko), c3);
      }
    }
  }
#pragma unroll
  for (int j = 0; j < 8; ++j) {
    const size_t ro = (size_t)(m0 + j + half * 8) * 4096 + n0 + colT;
    sums[ro] = c0[j];
    sums[ro + 16] = c1[j];
    sums[ro + 32] = c2[j];
    sums[ro + 48] = c3[j];
  }
}

// ---------------------------------------------------------------------------
// LSTM gate epilogue, mean over P=2.
// ---------------------------------------------------------------------------
__global__ __launch_bounds__(256) void cores_epi_kernel(
    const float* __restrict__ sums, const float* __restrict__ bi,
    const float* __restrict__ bh, const float* __restrict__ ba,
    const float* __restrict__ prev_c, float* __restrict__ out_c,
    float* __restrict__ nh32) {
  const int i = blockIdx.x * 256 + threadIdx.x;  // 0 .. B*R-1
  const int b = i >> 9, r = i & 511;
  const float pc = prev_c[i];
  float ac = 0.f, ah = 0.f;
#pragma unroll
  for (int p = 0; p < 2; ++p) {
    const float* s = sums + (size_t)b * 4096 + p * 2048;
    const int o = p * 2048;
    const float si = s[r] + bi[o + r] + bh[o + r] + ba[o + r];
    const float sf = s[512 + r] + bi[o + 512 + r] + bh[o + 512 + r] + ba[o + 512 + r];
    const float so = s[1024 + r] + bi[o + 1024 + r] + bh[o + 1024 + r] + ba[o + 1024 + r];
    const float st = s[1536 + r] + bi[o + 1536 + r] + bh[o + 1536 + r] + ba[o + 1536 + r];
    const float ig = 1.f / (1.f + __expf(-si));
    const float fg = 1.f / (1.f + __expf(-sf));
    const float og = 1.f / (1.f + __expf(-so));
    const float cp = fg * pc + ig * tanhf(st);
    ac += cp;
    ah += og * tanhf(cp);
  }
  out_c[i] = ac * 0.5f;
  nh32[i] = ah * 0.5f;
}

// ---------------------------------------------------------------------------
// top_h = att_res2 + next_h; also emit f16 copy and (layer0) xt1 = x + top_h.
// ---------------------------------------------------------------------------
__global__ __launch_bounds__(256) void toph_kernel(
    const float* __restrict__ ar2, const float* __restrict__ nh,
    const float* __restrict__ x, float* __restrict__ out_slot,
    _Float16* __restrict__ th16, _Float16* __restrict__ xt_next, int make_xt) {
  const int i = blockIdx.x * 256 + threadIdx.x;
  const float v = ar2[i] + nh[i];
  out_slot[i] = v;
  th16[i] = (_Float16)v;
  if (make_xt) xt_next[i] = (_Float16)(x[i] + v);
}

// ---------------------------------------------------------------------------
// Projection GEMM: logits[b,v] = top_h[b,:] . proj_w[v,:] + proj_b[v]
// M=1024, N=10000 (625 exact 16-tiles), K=512. grid (64, 20), waves own 4
// tiles. Edge tiles (>=625) are clamped; only the store is guarded.
// ---------------------------------------------------------------------------
__global__ __launch_bounds__(256) void proj_gemm_kernel(
    const _Float16* __restrict__ A, const _Float16* __restrict__ W,
    const float* __restrict__ pb, float* __restrict__ out) {
  const int wave = threadIdx.x >> 5, lane = threadIdx.x & 31;
  const int colT = lane & 15, half = lane >> 4;
  const int m0 = blockIdx.x * 16;
  const int ntb = blockIdx.y * 32 + wave * 4;
  const int aoff = (m0 + colT) * 512 + half * 16;

  const _Float16* Bp[4];
#pragma unroll
  for (int nt = 0; nt < 4; ++nt) {
    int tc = ntb + nt;
    if (tc > 624) tc = 624;  // clamp: wasted WMMA on <3% of waves, no branches
    Bp[nt] = W + (size_t)(tc * 16 + colT) * 512 + half * 16;
  }

  v8f c0 = {}, c1 = {}, c2 = {}, c3 = {};
#pragma unroll
  for (int kk = 0; kk < 2; ++kk) {
    v16h afr[8];
#pragma unroll
    for (int k = 0; k < 8; ++k) afr[k] = load_a16(A + aoff, (kk * 8 + k) * 32);
#pragma unroll
    for (int k = 0; k < 8; ++k) {
      const int ko = (kk * 8 + k) * 32;
      c0 = wmma16(afr[k], load_b16(Bp[0], ko), c0);
      c1 = wmma16(afr[k], load_b16(Bp[1], ko), c1);
      c2 = wmma16(afr[k], load_b16(Bp[2], ko), c2);
      c3 = wmma16(afr[k], load_b16(Bp[3], ko), c3);
    }
  }

  const v8f cc[4] = {c0, c1, c2, c3};
#pragma unroll
  for (int nt = 0; nt < 4; ++nt) {
    const int n0 = (ntb + nt) * 16;
    if (n0 >= 10000) continue;
    const int col = n0 + colT;
    const float bv = pb[col];
#pragma unroll
    for (int j = 0; j < 8; ++j)
      out[(size_t)(m0 + j + half * 8) * 10000 + col] = cc[nt][j] + bv;
  }
}

// ---------------------------------------------------------------------------
// In-place log_softmax over V=10000 per row.
// ---------------------------------------------------------------------------
__global__ __launch_bounds__(256) void lsm_kernel(float* __restrict__ out) {
  __shared__ float red[256];
  const int b = blockIdx.x, t = threadIdx.x;
  float* row = out + (size_t)b * 10000;
  float mx = -3.4e38f;
  for (int i = t; i < 10000; i += 256) mx = fmaxf(mx, row[i]);
  red[t] = mx;
  __syncthreads();
  for (int s = 128; s > 0; s >>= 1) {
    if (t < s) red[t] = fmaxf(red[t], red[t + s]);
    __syncthreads();
  }
  mx = red[0];
  __syncthreads();
  float sum = 0.f;
  for (int i = t; i < 10000; i += 256) sum += __expf(row[i] - mx);
  red[t] = sum;
  __syncthreads();
  for (int s = 128; s > 0; s >>= 1) {
    if (t < s) red[t] += red[t + s];
    __syncthreads();
  }
  const float lse = mx + logf(red[0]);
  __syncthreads();
  for (int i = t; i < 10000; i += 256) row[i] -= lse;
}

// ---------------------------------------------------------------------------
// Host launcher
// ---------------------------------------------------------------------------
extern "C" void kernel_launch(void* const* d_in, const int* in_sizes, int n_in,
                              void* d_out, int out_size, void* d_ws,
                              size_t ws_size, hipStream_t stream) {
  (void)in_sizes; (void)n_in; (void)out_size; (void)ws_size;

  constexpr int Bb = 1024, Aa = 196, Rr = 512, Vv = 10000;
  constexpr size_t BR = (size_t)Bb * Rr;            // 524288
  constexpr size_t BAR = (size_t)Bb * Aa * Rr;      // 102760448
  constexpr size_t AR = (size_t)Aa * Rr;            // 100352
  constexpr size_t CW = (size_t)2 * 4 * Rr * Rr;    // per-layer core W: 2097152
  constexpr size_t PW = (size_t)Vv * Rr;            // per-layer proj W: 5120000
  constexpr size_t BV = (size_t)Bb * Vv;            // 10240000

  const float* x      = (const float*)d_in[0];
  const float* att    = (const float*)d_in[1];
  const float* states = (const float*)d_in[2];
  const float* i2h_w  = (const float*)d_in[3];
  const float* i2h_b  = (const float*)d_in[4];
  const float* h2h_w  = (const float*)d_in[5];
  const float* h2h_b  = (const float*)d_in[6];
  const float* a2h_w  = (const float*)d_in[7];
  const float* a2h_b  = (const float*)d_in[8];
  const float* a2a_w  = (const float*)d_in[9];
  const float* a2a_b  = (const float*)d_in[10];
  const float* h2a_w  = (const float*)d_in[11];
  const float* h2a_b  = (const float*)d_in[12];
  const float* d2d_w  = (const float*)d_in[13];
  // d_in[14] = att_d2d_b: softmax shift-invariant -> unused
  const float* proj_w = (const float*)d_in[15];
  const float* proj_b = (const float*)d_in[16];
  float* out = (float*)d_out;

  // ---- workspace partition (~294 MB total) ----
  uintptr_t cur = (uintptr_t)d_ws;
  auto take = [&](size_t bytes) -> void* {
    cur = (cur + 255) & ~(uintptr_t)255;
    void* p = (void*)cur;
    cur += bytes;
    return p;
  };
  _Float16* att16  = (_Float16*)take(BAR * 2);
  _Float16* a2a16  = (_Float16*)take(4 * AR * 2);
  _Float16* i2h16  = (_Float16*)take(2 * CW * 2);
  _Float16* h2h16  = (_Float16*)take(2 * CW * 2);
  _Float16* a2h16  = (_Float16*)take(2 * CW * 2);
  _Float16* proj16 = (_Float16*)take(2 * PW * 2);
  _Float16* st16   = (_Float16*)take(4 * BR * 2);
  _Float16* x16    = (_Float16*)take(BR * 2);
  _Float16* xt116  = (_Float16*)take(BR * 2);
  float*    sums   = (float*)take((size_t)Bb * 4096 * 4);
  float*    ar32[4];
  _Float16* ar16[4];
  for (int i = 0; i < 4; ++i) ar32[i] = (float*)take(BR * 4);
  for (int i = 0; i < 4; ++i) ar16[i] = (_Float16*)take(BR * 2);
  float* nh32[2];
  for (int i = 0; i < 2; ++i) nh32[i] = (float*)take(BR * 4);
  _Float16* th16[2];
  for (int i = 0; i < 2; ++i) th16[i] = (_Float16*)take(BR * 2);

  // ---- f32 -> f16 conversions ----
  auto cvt = [&](const float* s, _Float16* d, size_t n) {
    cvt_f16_kernel<<<4096, 256, 0, stream>>>(s, d, (long long)(n / 4));
  };
  cvt(att, att16, BAR);
  cvt(a2a_w, a2a16, 4 * AR);
  cvt(i2h_w, i2h16, 2 * CW);
  cvt(h2h_w, h2h16, 2 * CW);
  cvt(a2h_w, a2h16, 2 * CW);
  cvt(proj_w, proj16, 2 * PW);
  cvt(states, st16, 4 * BR);
  cvt(x, x16, BR);

  auto soft_att = [&](int set, const float* h, float* o32, _Float16* o16) {
    soft_att_kernel<<<Bb, 256, SATT_LDS_BYTES, stream>>>(
        att16, att, h, a2a16 + set * AR, a2a_b + set * Aa, h2a_w + set * AR,
        h2a_b + set * Aa, d2d_w + set * Aa, o32, o16);
  };

  // att#1 for both layers (depend only on input states) -> can run up front
  soft_att(0, states + 1 * BR, ar32[0], ar16[0]);
  soft_att(2, states + 3 * BR, ar32[2], ar16[2]);

  for (int l = 0; l < 2; ++l) {
    const _Float16* xt = (l == 0) ? x16 : xt116;
    // cores GEMM + epilogue
    cores_gemm_kernel<<<dim3(64, 8), 256, 0, stream>>>(
        xt, st16 + (2 * l + 1) * BR, ar16[2 * l], i2h16 + l * CW,
        h2h16 + l * CW, a2h16 + l * CW, sums);
    cores_epi_kernel<<<(int)(BR / 256), 256, 0, stream>>>(
        sums, i2h_b + l * 4096, h2h_b + l * 4096, a2h_b + l * 4096,
        states + (2 * l) * BR, out + (2 * l) * BR, nh32[l]);
    // att#2 on next_h
    soft_att(2 * l + 1, nh32[l], ar32[2 * l + 1], ar16[2 * l + 1]);
    // top_h (+ xt for next layer)
    toph_kernel<<<(int)(BR / 256), 256, 0, stream>>>(
        ar32[2 * l + 1], nh32[l], x, out + (2 * l + 1) * BR, th16[l], xt116,
        (l == 0) ? 1 : 0);
    // projection + log_softmax (in place in d_out)
    float* logits = out + 4 * BR + (size_t)l * BV;
    proj_gemm_kernel<<<dim3(64, 20), 256, 0, stream>>>(
        th16[l], proj16 + l * PW, proj_b + l * Vv, logits);
    lsm_kernel<<<Bb, 256, 0, stream>>>(logits);
  }
}
